// GPT3_2422361555285
// MI455X (gfx1250) — matrix operand order repaired and checked by reading, not compiled, not run on hardware
//
#include <hip/hip_runtime.h>

typedef __attribute__((ext_vector_type(16))) __bf16 v16bf;
typedef __attribute__((ext_vector_type(8)))  __bf16 v8bf;
typedef __attribute__((ext_vector_type(8)))  float  v8f;
typedef __attribute__((ext_vector_type(4)))  int    v4i;

#define S_  1024
#define D_  1024
#define H_  16
#define HD_ 64
#define F_  4096
#define V_  32000
#define L_  2
#define TD_ 3072   // 3*D

#define AS1 __attribute__((address_space(1)))
#define AS3 __attribute__((address_space(3)))

#if defined(__AMDGCN__) && __has_builtin(__builtin_amdgcn_global_load_async_to_lds_b128) && __has_builtin(__builtin_amdgcn_s_wait_asynccnt)
#define USE_ASYNC_LDS 1
#define GL_ASYNC(gp, lp) \
  __builtin_amdgcn_global_load_async_to_lds_b128((AS1 v4i*)(gp), (AS3 v4i*)(lp), 0, 0)
#else
#define USE_ASYNC_LDS 0
#endif

// ------------------------------------------------------------- f32 -> bf16
__global__ __launch_bounds__(256)
void cvt_kernel(const float* __restrict__ src, __bf16* __restrict__ dst) {
  size_t i = ((size_t)blockIdx.x * 256 + threadIdx.x) * 8;
  float4 a = *(const float4*)(src + i);
  float4 b = *(const float4*)(src + i + 4);
  v8bf o;
  o[0] = (__bf16)a.x; o[1] = (__bf16)a.y; o[2] = (__bf16)a.z; o[3] = (__bf16)a.w;
  o[4] = (__bf16)b.x; o[5] = (__bf16)b.y; o[6] = (__bf16)b.z; o[7] = (__bf16)b.w;
  *(v8bf*)(dst + i) = o;
}

// ---------------------------------------------------------------- embedding
__global__ __launch_bounds__(256)
void embed_kernel(const int* __restrict__ tokens, const float* __restrict__ emb,
                  float* __restrict__ x) {
  int s = blockIdx.x;
  int t = tokens[s];
  for (int d = threadIdx.x; d < D_; d += 256)
    x[(size_t)s * D_ + d] = emb[(size_t)t * D_ + d];
}

// ---------------------------------------------------------------- layernorm
__global__ __launch_bounds__(256)
void ln_kernel(const float* __restrict__ x, const float* __restrict__ scale,
               const float* __restrict__ bias, float* __restrict__ y) {
  __shared__ float red[256];
  const int row = blockIdx.x, tid = threadIdx.x;
  const float* xr = x + (size_t)row * D_;
  float s = 0.f;
  for (int d = tid; d < D_; d += 256) s += xr[d];
  red[tid] = s; __syncthreads();
  for (int o = 128; o > 0; o >>= 1) { if (tid < o) red[tid] += red[tid + o]; __syncthreads(); }
  float mu = red[0] * (1.f / D_);
  __syncthreads();
  float v = 0.f;
  for (int d = tid; d < D_; d += 256) { float df = xr[d] - mu; v += df * df; }
  red[tid] = v; __syncthreads();
  for (int o = 128; o > 0; o >>= 1) { if (tid < o) red[tid] += red[tid + o]; __syncthreads(); }
  float inv = rsqrtf(red[0] * (1.f / D_) + 1e-6f);
  for (int d = tid; d < D_; d += 256)
    y[(size_t)row * D_ + d] = (xr[d] - mu) * inv * scale[d] + bias[d];
}

// -------------------------------- WMMA GEMM:  C = A(bf16) * W(bf16) + bias (+add)
// A: MxK bf16 (lda=K), W: KxN bf16 (ldb=N), C/bias/add: f32.
// 256 threads = 8 waves; block tile 128x64, BK=32; wave grid 4x2 of 32x32 quads.
// A tile: double-buffered LDS via GLOBAL_LOAD_ASYNC_TO_LDS_B128.
// B tile: software-pipelined (global b128 load issued before WMMAs, transposed
//         LDS store after), stored N-major so fragments are ds_load_b128.
template <bool HAS_ADD>
__global__ __launch_bounds__(256)
void gemm_kernel(const __bf16* __restrict__ A, const __bf16* __restrict__ W,
                 const float* __restrict__ bias, const float* __restrict__ add,
                 float* __restrict__ C, int M, int N, int K) {
  __shared__ __bf16 As[2][128][40];   // [m][k], 32 + 8 pad (80B row, 16B aligned)
  __shared__ __bf16 Bs[2][64][40];    // [n][k] transposed, 32 + 8 pad

  const int tid  = threadIdx.x;
  const int lane = tid & 31, wave = tid >> 5;
  const int wm = wave >> 1, wn = wave & 1;      // 4x2 waves over 128x64
  const int lm = lane & 15, lh = lane >> 4;
  const int m0 = blockIdx.y * 128;
  const int n0 = blockIdx.x * 64;

  // A: 512 16-byte chunks, 2 per thread
  const int ar0 = tid >> 2, ar1 = (tid + 256) >> 2, ac = (tid & 3) * 8;
  // B: 256 8-element chunks of a W row, 1 per thread
  const int br = tid >> 3, bc = (tid & 7) * 8;

  auto stageA = [&](int k0, int buf) {
#if USE_ASYNC_LDS
    GL_ASYNC(&A[(size_t)(m0 + ar0) * K + k0 + ac], &As[buf][ar0][ac]);
    GL_ASYNC(&A[(size_t)(m0 + ar1) * K + k0 + ac], &As[buf][ar1][ac]);
#else
    *(v8bf*)&As[buf][ar0][ac] = *(const v8bf*)&A[(size_t)(m0 + ar0) * K + k0 + ac];
    *(v8bf*)&As[buf][ar1][ac] = *(const v8bf*)&A[(size_t)(m0 + ar1) * K + k0 + ac];
#endif
  };
  auto loadB = [&](int k0) {
    return *(const v8bf*)&W[(size_t)(k0 + br) * N + n0 + bc];
  };
  auto storeB = [&](v8bf v, int buf) {   // transpose: row br of W -> column br
#pragma unroll
    for (int j = 0; j < 8; ++j) Bs[buf][bc + j][br] = v[j];
  };

  v8f acc[2][2] = {};

  stageA(0, 0);
  storeB(loadB(0), 0);
#if USE_ASYNC_LDS
  __builtin_amdgcn_s_wait_asynccnt(0);
#endif
  __syncthreads();

  for (int k0 = 0; k0 < K; k0 += 32) {
    const int  buf  = (k0 >> 5) & 1;
    const bool more = (k0 + 32) < K;

    v8bf bn = {};
    if (more) {
      stageA(k0 + 32, buf ^ 1);     // async fill of other A buffer
      bn = loadB(k0 + 32);          // global load in flight across the WMMAs
    }

    v16bf af[2], bfr[2];
#pragma unroll
    for (int t = 0; t < 2; ++t) {
      // A frag: af[0..7] = As[row][lh*8 ..], af[8..15] = As[row][16+lh*8 ..]
      int row = wm * 32 + t * 16 + lm;
      v8bf alo = *(const v8bf*)&As[buf][row][lh * 8];
      v8bf ahi = *(const v8bf*)&As[buf][row][16 + lh * 8];
      af[t] = __builtin_shufflevector(alo, ahi, 0, 1, 2, 3, 4, 5, 6, 7,
                                      8, 9, 10, 11, 12, 13, 14, 15);
      // B frag: bfr[0..7] = Bs[col][lh*8 ..], bfr[8..15] = Bs[col][16+lh*8 ..]
      int col = wn * 32 + t * 16 + lm;
      v8bf blo = *(const v8bf*)&Bs[buf][col][lh * 8];
      v8bf bhi = *(const v8bf*)&Bs[buf][col][16 + lh * 8];
      bfr[t] = __builtin_shufflevector(blo, bhi, 0, 1, 2, 3, 4, 5, 6, 7,
                                       8, 9, 10, 11, 12, 13, 14, 15);
    }
#pragma unroll
    for (int tm = 0; tm < 2; ++tm)
#pragma unroll
      for (int tn = 0; tn < 2; ++tn)
        acc[tm][tn] = __builtin_amdgcn_wmma_f32_16x16x32_bf16(
            false, af[tm], false, bfr[tn], (short)0, acc[tm][tn], false, false);

    if (more) storeB(bn, buf ^ 1);
#if USE_ASYNC_LDS
    __builtin_amdgcn_s_wait_asynccnt(0);
#endif
    __syncthreads();
  }

  // C/D layout: VGPR r -> M = r + lh*8, N = lm
#pragma unroll
  for (int tm = 0; tm < 2; ++tm) {
#pragma unroll
    for (int tn = 0; tn < 2; ++tn) {
      int n = n0 + wn * 32 + tn * 16 + lm;
      float bn = bias[n];
#pragma unroll
      for (int r = 0; r < 8; ++r) {
        int m = m0 + wm * 32 + tm * 16 + r + lh * 8;
        float val = acc[tm][tn][r] + bn;
        if (HAS_ADD) val += add[(size_t)m * N + n];
        C[(size_t)m * N + n] = val;
      }
    }
  }
}

// --------------------------------------------- attention (scores/softmax/ctx)
// Reference reshape is a flat re-view: flat f = h*65536 + k*64 + j maps to
// proj row (h*64 + (k>>4)), col ((k&15)*64 + j) -> each head-row is contiguous.
__global__ __launch_bounds__(256)
void attn_kernel(const float* __restrict__ proj, float* __restrict__ ctx,
                 float* __restrict__ attw) {
  const int p = blockIdx.x;   // query position
  const int h = blockIdx.y;   // head
  const int tid = threadIdx.x;
  __shared__ float qv[HD_];
  __shared__ float sc[S_];
  __shared__ float red[256];
  __shared__ float part[4][HD_];

  if (tid < HD_)
    qv[tid] = proj[(size_t)(h * 64 + (p >> 4)) * TD_ + (p & 15) * 64 + tid];
  __syncthreads();

  for (int k = tid; k < S_; k += 256) {
    float s = 0.f;
    if (k <= p) {
      const float4* kv = (const float4*)(proj + (size_t)(h * 64 + (k >> 4)) * TD_
                                               + D_ + (k & 15) * 64);
      const float4* q4 = (const float4*)qv;
      for (int j = 0; j < 16; ++j) {
        float4 a = q4[j], b = kv[j];
        s += a.x * b.x + a.y * b.y + a.z * b.z + a.w * b.w;
      }
      s *= (1.f / 32.f);  // 1/sqrt(D)
    }
    sc[k] = s;            // masked entries stay exactly 0 (mask multiplies)
  }
  __syncthreads();

  float mx = -3.4e38f;
  for (int k = tid; k < S_; k += 256) mx = fmaxf(mx, sc[k]);
  red[tid] = mx; __syncthreads();
  for (int o = 128; o > 0; o >>= 1) { if (tid < o) red[tid] = fmaxf(red[tid], red[tid + o]); __syncthreads(); }
  mx = red[0]; __syncthreads();

  float sm = 0.f;
  for (int k = tid; k < S_; k += 256) { float e = expf(sc[k] - mx); sc[k] = e; sm += e; }
  red[tid] = sm; __syncthreads();
  for (int o = 128; o > 0; o >>= 1) { if (tid < o) red[tid] += red[tid + o]; __syncthreads(); }
  float inv = 1.f / red[0];

  float* wrow = attw + ((size_t)h * S_ + p) * S_;
  for (int k = tid; k < S_; k += 256) { float w = sc[k] * inv; sc[k] = w; wrow[k] = w; }
  __syncthreads();

  // ctx[h][p][j] = sum_k w[k] * vh[h][k][j]; 64 dims x 4 key-chunks
  const int j = tid & 63, ch = tid >> 6;
  float a = 0.f;
  for (int k = ch; k < S_; k += 4)
    a += sc[k] * proj[(size_t)(h * 64 + (k >> 4)) * TD_ + 2 * D_ + (k & 15) * 64 + j];
  part[ch][j] = a;
  __syncthreads();
  if (tid < HD_)
    ctx[(size_t)h * (S_ * HD_) + (size_t)p * HD_ + tid] =
        part[0][tid] + part[1][tid] + part[2][tid] + part[3][tid];
}

// ---------------------------------------------------------------- gated GELU
__global__ __launch_bounds__(256)
void gated_kernel(const float* __restrict__ g, float* __restrict__ act) {
  size_t idx = (size_t)blockIdx.x * 256 + threadIdx.x;   // over S*F
  int s = (int)(idx >> 12);
  int f = (int)(idx & (F_ - 1));
  float a    = g[(size_t)s * (2 * F_) + f];
  float gate = g[(size_t)s * (2 * F_) + F_ + f];
  float t = tanhf(gate * 0.7978845608f * (1.f + 0.044715f * gate * gate));
  act[idx] = a * 0.5f * gate * (1.f + t);
}

// ---------------------------------------------------------------- z = x + y
__global__ __launch_bounds__(256)
void add_kernel(const float* __restrict__ x, const float* __restrict__ y,
                float* __restrict__ z) {
  size_t idx = (size_t)blockIdx.x * 256 + threadIdx.x;
  z[idx] = x[idx] + y[idx];
}

// ---------------------------------------------------------------------------
extern "C" void kernel_launch(void* const* d_in, const int* in_sizes, int n_in,
                              void* d_out, int out_size, void* d_ws, size_t ws_size,
                              hipStream_t stream) {
  const int*   tokens = (const int*)  d_in[0];
  const float* embw   = (const float*)d_in[1];
  const float* qkv1_w = (const float*)d_in[2];
  const float* qkv1_b = (const float*)d_in[3];
  const float* out1_w = (const float*)d_in[4];
  const float* out1_b = (const float*)d_in[5];
  const float* qkv2_w = (const float*)d_in[6];
  const float* qkv2_b = (const float*)d_in[7];
  const float* out2_w = (const float*)d_in[8];
  const float* out2_b = (const float*)d_in[9];
  const float* ln1_s  = (const float*)d_in[10];
  const float* ln1_b  = (const float*)d_in[11];
  const float* ln2_s  = (const float*)d_in[12];
  const float* ln2_b  = (const float*)d_in[13];
  const float* ln3_s  = (const float*)d_in[14];
  const float* ln3_b  = (const float*)d_in[15];
  const float* ffn1_w = (const float*)d_in[16];
  const float* ffn1_b = (const float*)d_in[17];
  const float* ffng_w = (const float*)d_in[18];
  const float* ffng_b = (const float*)d_in[19];
  const float* ffn2_w = (const float*)d_in[20];
  const float* ffn2_b = (const float*)d_in[21];
  const float* head_w = (const float*)d_in[22];
  const float* head_b = (const float*)d_in[23];

  float* out  = (float*)d_out;
  float* att1 = out + (size_t)S_ * V_;
  float* att2 = att1 + (size_t)L_ * H_ * S_ * S_;

  // ---- workspace layout ----
  char* wsb = (char*)d_ws;
  size_t off = 0;
  auto afl = [&](size_t n) { float* p = (float*)(wsb + off); off += n * sizeof(float); return p; };
  auto abf16 = [&](size_t n) { __bf16* p = (__bf16*)(wsb + off); off += n * sizeof(__bf16); return p; };

  float* x    = afl((size_t)S_ * D_);
  float* xn   = afl((size_t)S_ * D_);
  float* proj = afl((size_t)S_ * TD_);
  float* ctx  = afl((size_t)S_ * D_);
  float* a2   = afl((size_t)S_ * D_);
  float* hb   = afl((size_t)S_ * F_);
  float* gb   = afl((size_t)S_ * 2 * F_);
  float* act  = afl((size_t)S_ * F_);
  __bf16* abf = abf16((size_t)S_ * F_);            // shared bf16 activation buffer

  __bf16* w_qkv1 = abf16((size_t)L_ * D_ * TD_);
  __bf16* w_out1 = abf16((size_t)L_ * D_ * D_);
  __bf16* w_qkv2 = abf16((size_t)L_ * D_ * TD_);
  __bf16* w_out2 = abf16((size_t)L_ * D_ * D_);
  __bf16* w_ff1  = abf16((size_t)L_ * D_ * F_);
  __bf16* w_ffg  = abf16((size_t)L_ * F_ * 2 * F_);
  __bf16* w_ff2  = abf16((size_t)L_ * F_ * D_);
  __bf16* w_head = abf16((size_t)D_ * V_);

  auto cvt = [&](const float* s, __bf16* d, size_t n) {
    cvt_kernel<<<(unsigned)(n / 2048), 256, 0, stream>>>(s, d);
  };
  auto gemm = [&](const __bf16* A, const __bf16* W, const float* b,
                  const float* add, float* C, int M, int N, int K) {
    dim3 grid(N / 64, M / 128);
    if (add)
      gemm_kernel<true><<<grid, 256, 0, stream>>>(A, W, b, add, C, M, N, K);
    else
      gemm_kernel<false><<<grid, 256, 0, stream>>>(A, W, b, add, C, M, N, K);
  };

  // ---- one-time (per launch) weight down-conversion to bf16 ----
  cvt(qkv1_w, w_qkv1, (size_t)L_ * D_ * TD_);
  cvt(out1_w, w_out1, (size_t)L_ * D_ * D_);
  cvt(qkv2_w, w_qkv2, (size_t)L_ * D_ * TD_);
  cvt(out2_w, w_out2, (size_t)L_ * D_ * D_);
  cvt(ffn1_w, w_ff1,  (size_t)L_ * D_ * F_);
  cvt(ffng_w, w_ffg,  (size_t)L_ * F_ * 2 * F_);
  cvt(ffn2_w, w_ff2,  (size_t)L_ * F_ * D_);
  cvt(head_w, w_head, (size_t)D_ * V_);

  embed_kernel<<<S_, 256, 0, stream>>>(tokens, embw, x);

  for (int l = 0; l < L_; ++l) {
    // x = LN1(x); x = x + attn1(x)
    ln_kernel<<<S_, 256, 0, stream>>>(x, ln1_s + l * D_, ln1_b + l * D_, xn);
    cvt(xn, abf, (size_t)S_ * D_);
    gemm(abf, w_qkv1 + (size_t)l * D_ * TD_, qkv1_b + (size_t)l * TD_, nullptr,
         proj, S_, TD_, D_);
    attn_kernel<<<dim3(S_, H_), 256, 0, stream>>>(
        proj, ctx, att1 + (size_t)l * H_ * S_ * S_);
    cvt(ctx, abf, (size_t)S_ * D_);
    gemm(abf, w_out1 + (size_t)l * D_ * D_, out1_b + (size_t)l * D_, xn,
         x, S_, D_, D_);

    // x = LN2(x); a2 = attn2(x); x = x + a2
    ln_kernel<<<S_, 256, 0, stream>>>(x, ln2_s + l * D_, ln2_b + l * D_, xn);
    cvt(xn, abf, (size_t)S_ * D_);
    gemm(abf, w_qkv2 + (size_t)l * D_ * TD_, qkv2_b + (size_t)l * TD_, nullptr,
         proj, S_, TD_, D_);
    attn_kernel<<<dim3(S_, H_), 256, 0, stream>>>(
        proj, ctx, att2 + (size_t)l * H_ * S_ * S_);
    cvt(ctx, abf, (size_t)S_ * D_);
    gemm(abf, w_out2 + (size_t)l * D_ * D_, out2_b + (size_t)l * D_, nullptr,
         a2, S_, D_, D_);
    add_kernel<<<(S_ * D_) / 256, 256, 0, stream>>>(xn, a2, x);

    // ff = FFN(LN3(x)); x = ff + a2
    ln_kernel<<<S_, 256, 0, stream>>>(x, ln3_s + l * D_, ln3_b + l * D_, xn);
    cvt(xn, abf, (size_t)S_ * D_);
    gemm(abf, w_ff1 + (size_t)l * D_ * F_, ffn1_b + (size_t)l * F_, nullptr,
         hb, S_, F_, D_);
    cvt(hb, abf, (size_t)S_ * F_);
    gemm(abf, w_ffg + (size_t)l * F_ * 2 * F_, ffng_b + (size_t)l * 2 * F_,
         nullptr, gb, S_, 2 * F_, F_);
    gated_kernel<<<(S_ * F_) / 256, 256, 0, stream>>>(gb, act);
    cvt(act, abf, (size_t)S_ * F_);
    gemm(abf, w_ff2 + (size_t)l * F_ * D_, ffn2_b + (size_t)l * D_, a2,
         x, S_, D_, F_);
  }

  // logits = x @ head_w + head_b
  cvt(x, abf, (size_t)S_ * D_);
  gemm(abf, w_head, head_b, nullptr, out, S_, V_, D_);
}
